// Quantize_No_Transformer_41540923687461
// MI455X (gfx1250) — matrix operand order repaired and checked by reading, not compiled, not run on hardware
//
#include <hip/hip_runtime.h>
#include <stdint.h>
#include <math.h>

typedef __attribute__((ext_vector_type(16))) _Float16     v16h;
typedef __attribute__((ext_vector_type(8)))  float        v8f;
typedef __attribute__((ext_vector_type(4)))  unsigned int v4u;
typedef __attribute__((ext_vector_type(4)))  unsigned int u32x4;
typedef __attribute__((ext_vector_type(8)))  int          i32x8;
typedef __attribute__((ext_vector_type(4)))  int          i32x4;

#define B_ 8
#define T_ 4096
#define D_ 256
#define K_ 4096
#define N_ (B_*T_)              // 32768
#define DECAYF 0.8f
#define OMDECAYF 0.2f
#define COMMITF 1.0f
#define EPSF 1e-5f

#if defined(__has_builtin)
#if __has_builtin(__builtin_amdgcn_tensor_load_to_lds) && __has_builtin(__builtin_amdgcn_s_wait_tensorcnt)
#define HAVE_TDM 1
#endif
#endif
#ifndef HAVE_TDM
#define HAVE_TDM 0
#endif

// ---------------- workspace layout (float offsets) ----------------
static constexpr size_t F_CSTAT  = 0;                       // 4 x 2048 partial sums
static constexpr size_t F_STATS  = 8192;                    // cm, inv_cs, sm, ss
static constexpr size_t F_QSTAT  = 16384;                   // qsum, qsum2
static constexpr size_t F_SCAL   = 20480;                   // [0]=percep [1]=style [2]=n
static constexpr size_t F_ENORM  = 20544;                   // K
static constexpr size_t F_COUNTS = 24640;                   // K
static constexpr size_t F_ESUMT  = 28736;                   // K*D embed_sum^T (K,D)
static constexpr size_t F_EMBTF  = F_ESUMT + (size_t)K_*D_; // K*D fp32 codebook^T
static constexpr size_t F_TGT    = F_EMBTF + (size_t)K_*D_; // N*D fp32 target
static constexpr size_t F_END    = F_TGT + (size_t)N_*D_;
static constexpr size_t BY_TGT_H = F_END * 4;                          // N*D f16
static constexpr size_t BY_EMB_H = BY_TGT_H + (size_t)N_*D_*2;         // K*D f16
static constexpr size_t ZERO_FLOATS = F_ESUMT + (size_t)K_*D_;

union Frag { v16h v; v4u u[2]; };

__device__ __forceinline__ float waveSum(float v) {
    #pragma unroll
    for (int m = 16; m >= 1; m >>= 1) v += __shfl_xor(v, m, 32);
    return v;
}

// ---------------- kernels ----------------
__global__ void k_zero(float* p, size_t n) {
    for (size_t i = (size_t)blockIdx.x*blockDim.x + threadIdx.x; i < n;
         i += (size_t)gridDim.x*blockDim.x) p[i] = 0.0f;
}

__global__ __launch_bounds__(256) void k_stats_part(const float* __restrict__ cF,
                                                    const float* __restrict__ sF,
                                                    float* __restrict__ cstat) {
    int b = blockIdx.x >> 6, ch = blockIdx.x & 63, d = threadIdx.x;
    size_t base = ((size_t)b*T_ + (size_t)ch*64)*D_ + d;
    float s0=0.f, s1=0.f, s2=0.f, s3=0.f;
    for (int i = 0; i < 64; ++i) {
        float v = cF[base + (size_t)i*D_]; s0 += v; s1 += v*v;
        float w = sF[base + (size_t)i*D_]; s2 += w; s3 += w*w;
    }
    int o = b*D_ + d;
    atomicAdd(cstat + o,        s0); atomicAdd(cstat + 2048 + o, s1);
    atomicAdd(cstat + 4096 + o, s2); atomicAdd(cstat + 6144 + o, s3);
}

__global__ void k_fin_stats(const float* __restrict__ cstat, float* __restrict__ stats) {
    int i = blockIdx.x*blockDim.x + threadIdx.x;
    if (i >= B_*D_) return;
    float m   = cstat[i] * (1.0f/T_);
    float var = (cstat[2048+i] - (float)T_*m*m) * (1.0f/(T_-1));
    stats[i]      = m;
    stats[2048+i] = 1.0f / sqrtf(var + EPSF);      // 1/cs
    float m2   = cstat[4096+i] * (1.0f/T_);
    float var2 = (cstat[6144+i] - (float)T_*m2*m2) * (1.0f/(T_-1));
    stats[4096+i] = m2;                            // sm (== tm)
    stats[6144+i] = sqrtf(var2 + EPSF);            // ss (== ts)
}

__global__ void k_target(const float* __restrict__ cF, const float* __restrict__ stats,
                         float* __restrict__ tgtF, _Float16* __restrict__ tgtH) {
    size_t stride = (size_t)gridDim.x * blockDim.x;
    for (size_t i = (size_t)blockIdx.x*blockDim.x + threadIdx.x; i < (size_t)N_*D_; i += stride) {
        int b = (int)(i / ((size_t)T_*D_));
        int d = (int)(i & (D_-1));
        int o = b*D_ + d;
        float v = (cF[i] - stats[o]) * stats[2048+o] * stats[6144+o] + stats[4096+o];
        tgtF[i] = v;
        tgtH[i] = (_Float16)v;
    }
}

__global__ void k_transpose(const float* __restrict__ embed,
                            float* __restrict__ embTF, _Float16* __restrict__ embTH) {
    int i = blockIdx.x*256 + threadIdx.x;            // i = d*K + k
    if (i >= D_*K_) return;
    int d = i >> 12, k = i & (K_-1);
    float v = embed[i];
    embTF[(size_t)k*D_ + d] = v;
    embTH[(size_t)k*D_ + d] = (_Float16)v;
}

__global__ __launch_bounds__(256) void k_enorm(const float* __restrict__ embTF,
                                               float* __restrict__ enorm) {
    int wave = threadIdx.x >> 5, lane = threadIdx.x & 31;
    int k = blockIdx.x*8 + wave;
    float s = 0.f;
    for (int j = lane; j < D_; j += 32) { float v = embTF[(size_t)k*D_ + j]; s += v*v; }
    s = waveSum(s);
    if (lane == 0) enorm[k] = s;
}

static constexpr int LDA   = 264;                    // padded halves per row (bank spread)
static constexpr int SLAB  = 64;                     // codebook columns per slab
static constexpr int NSLAB = K_ / SLAB;

#if HAVE_TDM
// Issue a TDM load of one codebook slab (64 rows x 256 f16, row-major contiguous)
// into LDS with hardware padding of 4 DWORDs every 128 DWORDs -> 264-half LDS stride.
__device__ __forceinline__ void tdm_load_b_slab(const _Float16* gsrc, unsigned lds_off) {
    uint64_t ga = (uint64_t)(uintptr_t)gsrc;
    u32x4 g0;
    g0.x = 1u;                                              // count=1 (valid), user mode
    g0.y = lds_off;                                         // LDS byte address
    g0.z = (unsigned)(ga & 0xFFFFFFFFu);                    // global_addr[31:0]
    g0.w = (unsigned)((ga >> 32) & 0x1FFFFFFu) | (2u << 30);// global_addr[56:32] | type=2
    i32x8 g1;
    g1[0] = (1 << 16)        // data_size = 2 bytes
          | (1 << 20)        // pad_enable
          | (6 << 22)        // pad_interval: 128 DWORDs (=512B row)
          | (3 << 25);       // pad_amount: 4 DWORDs (=16B)
    g1[1] = (int)(256u << 16);   // tensor_dim0[15:0] = 256 @ bits[63:48]
    g1[2] = (int)(4096u << 16);  // tensor_dim1[15:0] = 4096 @ bits[95:80]
    g1[3] = (int)(256u << 16);   // tile_dim0 = 256 @ bits[127:112]
    g1[4] = SLAB;                // tile_dim1 = 64 @ bits[143:128]
    g1[5] = 256;                 // tensor_dim0_stride[31:0] = 256 elements
    g1[6] = 0;
    g1[7] = 0;
    i32x4 z4 = {0, 0, 0, 0};
    i32x8 z8 = {0, 0, 0, 0, 0, 0, 0, 0};
    __builtin_amdgcn_tensor_load_to_lds(g0, g1, z4, z4, z8, 0);
}
#endif

// Fused distance-GEMM + argmin + gather quantize + scatter embed_sum.
// Block = 256 (8 waves): 32 token rows; loop all K codes in 64-col slabs with a
// TDM-fed double-buffered LDS B tile.
__global__ __launch_bounds__(256) void k_argmin_gemm(
        const _Float16* __restrict__ tgtH, const _Float16* __restrict__ embTH,
        const float* __restrict__ embTF, const float* __restrict__ tgtF,
        const float* __restrict__ enorm,
        float* __restrict__ outQ, float* __restrict__ outInd,
        float* __restrict__ counts, float* __restrict__ esumT) {
    __shared__ _Float16 ldsA[32*LDA];
    __shared__ _Float16 ldsB[2][SLAB*LDA];
    __shared__ float resV[32*4];
    __shared__ int   resI[32*4];
    __shared__ int   kminS[32];

    const int tid  = threadIdx.x;
    const int lane = tid & 31;
    const int wave = tid >> 5;
    const int rowbase = blockIdx.x * 32;

    // stage 32x256 f16 A-tile into LDS (64B per thread, b128 moves)
    {
        int row = tid >> 3, seg = tid & 7;
        const v4u* g = reinterpret_cast<const v4u*>(tgtH + ((size_t)(rowbase+row))*D_ + seg*32);
        v4u* l = reinterpret_cast<v4u*>(&ldsA[row*LDA + seg*32]);
        l[0]=g[0]; l[1]=g[1]; l[2]=g[2]; l[3]=g[3];
    }

    const int rt   = wave & 1;        // row tile 0/1
    const int ct   = wave >> 1;       // col tile 0..3
    const int half = lane >> 4;       // half-wave
    const int l15  = lane & 15;
    const int arow   = rt*16 + l15;   // A row held by this lane
    const int khalfA = half * 8;      // A K-interleave (halves)
    const int koffB  = half * 8;      // B K offset (halves)

#if HAVE_TDM
    if (wave == 0) {
        tdm_load_b_slab(embTH, (unsigned)(uintptr_t)(void*)&ldsB[0][0]);
        __builtin_amdgcn_s_wait_tensorcnt(0);
    }
#else
    {   // cooperative fallback: slab 0 -> buffer 0
        int row = tid >> 2, seg = tid & 3;
        const v4u* g = reinterpret_cast<const v4u*>(embTH + (size_t)row*D_ + seg*64);
        v4u* l = reinterpret_cast<v4u*>(&ldsB[0][row*LDA + seg*64]);
        l[0]=g[0]; l[1]=g[1]; l[2]=g[2]; l[3]=g[3];
    }
#endif
    __syncthreads();

    // hoist the 8 A fragments for this lane into registers
    Frag afr[8];
    #pragma unroll
    for (int j = 0; j < 8; ++j) {
        const v4u* pa0 = reinterpret_cast<const v4u*>(&ldsA[arow*LDA + j*32 + khalfA]);
        const v4u* pa1 = reinterpret_cast<const v4u*>(&ldsA[arow*LDA + j*32 + 16 + khalfA]);
        afr[j].u[0] = pa0[0];
        afr[j].u[1] = pa1[0];
    }

    float minv[8]; int mini[8];
    #pragma unroll
    for (int r = 0; r < 8; ++r) { minv[r] = 3.4e38f; mini[r] = 0; }

    for (int s = 0; s < NSLAB; ++s) {
        const int p = s & 1;
#if HAVE_TDM
        if (wave == 0 && s + 1 < NSLAB)
            tdm_load_b_slab(embTH + (size_t)(s+1)*SLAB*D_,
                            (unsigned)(uintptr_t)(void*)&ldsB[(s+1)&1][0]);
#else
        if (s + 1 < NSLAB) {
            int row = tid >> 2, seg = tid & 3;
            const v4u* g = reinterpret_cast<const v4u*>(
                embTH + (size_t)((s+1)*SLAB + row)*D_ + seg*64);
            v4u* l = reinterpret_cast<v4u*>(&ldsB[(s+1)&1][row*LDA + seg*64]);
            l[0]=g[0]; l[1]=g[1]; l[2]=g[2]; l[3]=g[3];
        }
#endif
        const _Float16* ldsBp = &ldsB[p][0];
        const int bcol = ct*16 + l15;           // column within slab for this lane
        const int col  = s*SLAB + bcol;         // global code index

        // batch all 16 ds_load_b128 B-fragment reads, then 8 back-to-back WMMAs
        Frag bfr[8];
        #pragma unroll
        for (int j = 0; j < 8; ++j) {
            const v4u* pb = reinterpret_cast<const v4u*>(&ldsBp[bcol*LDA + j*32 + koffB]);
            bfr[j].u[0] = pb[0];
            bfr[j].u[1] = pb[2];
        }
        v8f c = {0.f,0.f,0.f,0.f,0.f,0.f,0.f,0.f};
        #pragma unroll
        for (int j = 0; j < 8; ++j)
            c = __builtin_amdgcn_wmma_f32_16x16x32_f16(false, afr[j].v, false, bfr[j].v,
                                                       (short)0, c, false, false);
        const float en = enorm[col];
        #pragma unroll
        for (int r = 0; r < 8; ++r) {
            float dist = en - 2.0f * c[r];      // ||f||^2 constant per row: omitted
            if (dist < minv[r]) { minv[r] = dist; mini[r] = col; }
        }
#if HAVE_TDM
        if (wave == 0 && s + 1 < NSLAB)
            __builtin_amdgcn_s_wait_tensorcnt(0);
#endif
        __syncthreads();
    }

    // argmin across the 16 column-lanes of each half-wave
    #pragma unroll
    for (int m = 8; m >= 1; m >>= 1) {
        #pragma unroll
        for (int r = 0; r < 8; ++r) {
            float ov = __shfl_xor(minv[r], m, 32);
            int   oi = __shfl_xor(mini[r], m, 32);
            if (ov < minv[r] || (ov == minv[r] && oi < mini[r])) { minv[r]=ov; mini[r]=oi; }
        }
    }
    if (l15 == 0) {
        #pragma unroll
        for (int r = 0; r < 8; ++r) {
            int row = rt*16 + half*8 + r;       // C layout: lanes16-31 hold rows +8
            resV[row*4 + ct] = minv[r];
            resI[row*4 + ct] = mini[r];
        }
    }
    __syncthreads();
    if (tid < 32) {
        float mv = resV[tid*4]; int mi = resI[tid*4];
        #pragma unroll
        for (int j = 1; j < 4; ++j) {
            float ov = resV[tid*4+j]; int oi = resI[tid*4+j];
            if (ov < mv || (ov == mv && oi < mi)) { mv = ov; mi = oi; }
        }
        kminS[tid] = mi;
        outInd[rowbase + tid] = (float)mi;
        atomicAdd(&counts[mi], 1.0f);
    }
    __syncthreads();
    // gather quantize rows + scatter embed_sum (coalesced transposed accumulator)
    for (int e = tid; e < 32*D_; e += 256) {
        int row = e >> 8, d = e & (D_-1);
        int k = kminS[row];
        size_t gi = (size_t)(rowbase + row)*D_ + d;
        outQ[gi] = embTF[(size_t)k*D_ + d];
        atomicAdd(&esumT[(size_t)k*D_ + d], tgtF[gi]);
    }
}

__global__ void k_ema_cluster(const float* __restrict__ cluster, const float* __restrict__ counts,
                              float* __restrict__ outCluster, float* __restrict__ scal) {
    int i = blockIdx.x*256 + threadIdx.x;
    float nc = 0.f;
    if (i < K_) { nc = cluster[i]*DECAYF + OMDECAYF*counts[i]; outCluster[i] = nc; }
    float s = waveSum(nc);
    if ((threadIdx.x & 31) == 0) atomicAdd(&scal[2], s);
}

__global__ void k_ema_avg(const float* __restrict__ embed_avg, const float* __restrict__ esumT,
                          float* __restrict__ outAvg) {
    int i = blockIdx.x*256 + threadIdx.x;             // i = d*K + k
    if (i >= D_*K_) return;
    int d = i >> 12, k = i & (K_-1);
    outAvg[i] = embed_avg[i]*DECAYF + OMDECAYF*esumT[(size_t)k*D_ + d];
}

__global__ void k_new_embed(const float* __restrict__ outAvg, const float* __restrict__ outCluster,
                            const float* __restrict__ scal, float* __restrict__ outEmbed) {
    int i = blockIdx.x*256 + threadIdx.x;
    if (i >= D_*K_) return;
    int k = i & (K_-1);
    float n = scal[2];
    float sm = (outCluster[k] + EPSF) / (n + (float)K_*EPSF) * n;
    outEmbed[i] = outAvg[i] / sm;
}

__global__ void k_percep(const float* __restrict__ q, const float* __restrict__ tgt,
                         float* __restrict__ scal) {
    float s = 0.f;
    size_t stride = (size_t)gridDim.x * blockDim.x;
    for (size_t i = (size_t)blockIdx.x*blockDim.x + threadIdx.x; i < (size_t)N_*D_; i += stride) {
        float d = q[i] - tgt[i]; s += d*d;
    }
    s = waveSum(s);
    if ((threadIdx.x & 31) == 0) atomicAdd(&scal[0], s);
}

__global__ __launch_bounds__(256) void k_qstats_part(const float* __restrict__ q,
                                                     float* __restrict__ qstat) {
    int b = blockIdx.x >> 6, ch = blockIdx.x & 63, d = threadIdx.x;
    size_t base = ((size_t)b*T_ + (size_t)ch*64)*D_ + d;
    float s0 = 0.f, s1 = 0.f;
    for (int i = 0; i < 64; ++i) { float v = q[base + (size_t)i*D_]; s0 += v; s1 += v*v; }
    int o = b*D_ + d;
    atomicAdd(qstat + o, s0); atomicAdd(qstat + 2048 + o, s1);
}

__global__ void k_style(const float* __restrict__ qstat, const float* __restrict__ stats,
                        float* __restrict__ scal) {
    int i = blockIdx.x*256 + threadIdx.x;
    float acc = 0.f;
    if (i < B_*D_) {
        float qm = qstat[i] * (1.0f/T_);
        float qv = (qstat[2048+i] - (float)T_*qm*qm) * (1.0f/(T_-1));
        float qs = sqrtf(qv + EPSF);
        float tm = stats[4096+i], ts = stats[6144+i];
        acc = (qm-tm)*(qm-tm) + (qs-ts)*(qs-ts);
    }
    acc = waveSum(acc);
    if ((threadIdx.x & 31) == 0) atomicAdd(&scal[1], acc);
}

__global__ void k_loss(const float* __restrict__ scal, float* __restrict__ outLoss) {
    float percep = scal[0] / ((float)N_ * (float)D_);
    float style  = scal[1] / ((float)B_ * (float)D_);
    outLoss[0] = percep * COMMITF + 5.0f * style;
}

// ---------------- host launch ----------------
extern "C" void kernel_launch(void* const* d_in, const int* in_sizes, int n_in,
                              void* d_out, int out_size, void* d_ws, size_t ws_size,
                              hipStream_t stream) {
    const float* cF        = (const float*)d_in[0];
    const float* sF        = (const float*)d_in[1];
    const float* embed     = (const float*)d_in[2];
    const float* cluster   = (const float*)d_in[3];
    const float* embed_avg = (const float*)d_in[4];

    float* out        = (float*)d_out;
    float* outQ       = out;                                  // (B,T,D)
    float* outInd     = outQ + (size_t)N_*D_;                 // (B,T) as float
    float* outLoss    = outInd + N_;                          // scalar
    float* outEmbed   = outLoss + 1;                          // (D,K)
    float* outCluster = outEmbed + (size_t)D_*K_;             // (K,)
    float* outAvg     = outCluster + K_;                      // (D,K)

    float*    wsf   = (float*)d_ws;
    float*    esumT = wsf + F_ESUMT;
    float*    embTF = wsf + F_EMBTF;
    float*    tgtF  = wsf + F_TGT;
    _Float16* tgtH  = (_Float16*)((char*)d_ws + BY_TGT_H);
    _Float16* embTH = (_Float16*)((char*)d_ws + BY_EMB_H);

    k_zero<<<2048, 256, 0, stream>>>(wsf, ZERO_FLOATS);
    k_stats_part<<<B_*64, 256, 0, stream>>>(cF, sF, wsf + F_CSTAT);
    k_fin_stats<<<8, 256, 0, stream>>>(wsf + F_CSTAT, wsf + F_STATS);
    k_target<<<8192, 256, 0, stream>>>(cF, wsf + F_STATS, tgtF, tgtH);
    k_transpose<<<4096, 256, 0, stream>>>(embed, embTF, embTH);
    k_enorm<<<K_/8, 256, 0, stream>>>(embTF, wsf + F_ENORM);
    k_argmin_gemm<<<N_/32, 256, 0, stream>>>(tgtH, embTH, embTF, tgtF, wsf + F_ENORM,
                                             outQ, outInd, wsf + F_COUNTS, esumT);
    k_ema_cluster<<<K_/256, 256, 0, stream>>>(cluster, wsf + F_COUNTS, outCluster, wsf + F_SCAL);
    k_ema_avg<<<(D_*K_)/256, 256, 0, stream>>>(embed_avg, esumT, outAvg);
    k_new_embed<<<(D_*K_)/256, 256, 0, stream>>>(outAvg, outCluster, wsf + F_SCAL, outEmbed);
    k_percep<<<4096, 256, 0, stream>>>(outQ, tgtF, wsf + F_SCAL);
    k_qstats_part<<<B_*64, 256, 0, stream>>>(outQ, wsf + F_QSTAT);
    k_style<<<8, 256, 0, stream>>>(wsf + F_QSTAT, wsf + F_STATS, wsf + F_SCAL);
    k_loss<<<1, 1, 0, stream>>>(wsf + F_SCAL, outLoss);
}
